// SNNDeep_678604833396
// MI455X (gfx1250) — matrix operand order repaired and checked by reading, not compiled, not run on hardware
//
#include <hip/hip_runtime.h>
#include <hip/hip_bf16.h>

typedef __attribute__((ext_vector_type(16))) _Float16 v16h;
typedef __attribute__((ext_vector_type(8)))  _Float16 v8h;
typedef __attribute__((ext_vector_type(8)))  float    v8f;

#define T_STEPS 200
#define BATCH   256
#define IN_F    512
#define H1      1024
#define H2      1024
#define H3      512

#define BETA1 0.8187307530779818f  // exp(-1/5)
#define BETA2 0.9048374180359595f  // exp(-1/10)
#define BETA3 0.9512294245007139f  // exp(-1/20)

// ---------------------------------------------------------------------------
// Pack f32 row-major weights W[Hout][K] into WMMA B-fragment-major f16:
// tile (nt,kt) covers N = nt*16..+15, K = kt*32..+31.
// B-fragment per-lane layout (16-bit B, 32x16): lane l holds column n = l&15,
// k = kt*32 + 8*(l>>4) + (i&7) + 16*(i>>3)   (i = 0..15 within the lane's 16 f16 values).
// Storage: dst[((nt*Ktiles + kt)*32 + lane)*16 + i]  -> one contiguous 32B
// v16h load per lane in the GEMM.
// ---------------------------------------------------------------------------
__global__ void snn_pack_w(const float* __restrict__ W,
                           _Float16* __restrict__ dst,
                           int Hout, int K) {
    int gid = blockIdx.x * blockDim.x + threadIdx.x;
    int total = Hout * K;
    if (gid >= total) return;
    int i    = gid & 15;
    int lane = (gid >> 4) & 31;
    int tile = gid >> 9;                 // 512 elements per 16x32 tile
    int ktiles = K >> 5;
    int nt = tile / ktiles;
    int kt = tile - nt * ktiles;
    int n = nt * 16 + (lane & 15);
    int k = kt * 32 + ((lane >> 4) << 3) + (i & 7) + ((i >> 3) << 4);
    dst[gid] = (_Float16)W[(size_t)n * K + k];
}

// ---------------------------------------------------------------------------
// Fused GEMM (WMMA f16 -> f32) + LIF update, one layer, one timestep.
//   cur   = act @ W^T + bias        (act = 0/1 spikes, f32 or f16 row-major)
//   reset = (mem_prev > 1);  mem = beta*mem_prev + cur - reset
//   spk   = (mem > 1)
//
// Block = 8 waves sharing one batch-tile mt. The 16xK A row-block is staged
// once into LDS in per-lane A-fragment order (f32 spikes converted to f16
// during staging), then each wave computes a 16x32 output tile (two n-tiles
// sharing one A fragment -> 2 WMMAs per k-tile) with pure ds_load_b128 A
// reads and contiguous 32B global B reads.
// A-fragment ISA layout (16-bit A 16x32): lane l holds row m = l&15,
// k = 8*(l>>4) + {0..7, 16..23}.
// ---------------------------------------------------------------------------
template<bool AF32, int K, int H, bool HAS_SPK, bool HAS_SUM>
__global__ __launch_bounds__(256) void snn_lif(
    const void*     __restrict__ actv,     // [BATCH, K] spikes (f32 or f16)
    const _Float16* __restrict__ Wp,       // packed weights (fragment-major)
    const float*    __restrict__ bias,
    float*          __restrict__ mem,      // [BATCH, H] membrane state
    _Float16*       __restrict__ spkOut,   // [BATCH, H] f16 spikes (if HAS_SPK)
    float*          __restrict__ memSum,   // [BATCH, H] running sum (if HAS_SUM)
    float*          __restrict__ spikeAcc, // 1 float: total spike count
    float beta)
{
    constexpr int KT  = K / 32;            // k-tiles
    constexpr int NP  = H / 32;            // n-tile pairs (multiple of 8)
    constexpr int NPB = NP / 8;            // blocks per batch-tile

    const int tid  = threadIdx.x;
    const int lane = tid & 31;
    const int mt   = blockIdx.x / NPB;                       // shared by block
    const int np   = (blockIdx.x - mt * NPB) * 8 + (tid >> 5);
    const int nt0  = np * 2;
    const int half = lane >> 4;
    const int ln   = lane & 15;

    // ---- stage A row-block into LDS, fragment-packed: sA[(kt*32+l)*16 + i]
    __shared__ _Float16 sA[16 * K];
    const float*    aF = (const float*)actv;
    const _Float16* aH = (const _Float16*)actv;
    constexpr int PAIRS = KT * 32;         // == K; multiple of 256
    constexpr int PER_T = PAIRS / 256;
#pragma unroll
    for (int p = 0; p < PER_T; ++p) {
        const int pair = tid + p * 256;
        const int l    = pair & 31;
        const int kt   = pair >> 5;
        const size_t src = (size_t)(mt * 16 + (l & 15)) * K
                         + kt * 32 + 8 * (l >> 4);
        v8h lo, hi;
        if constexpr (AF32) {
#pragma unroll
            for (int i = 0; i < 8; ++i) {
                lo[i] = (_Float16)aF[src + i];
                hi[i] = (_Float16)aF[src + 16 + i];
            }
        } else {
            lo = *(const v8h*)(aH + src);
            hi = *(const v8h*)(aH + src + 16);
        }
        *(v8h*)(&sA[pair * 16])     = lo;
        *(v8h*)(&sA[pair * 16 + 8]) = hi;
    }
    __syncthreads();

    // ---- accumulators initialized with the bias of their output column
    v8f c[2];
#pragma unroll
    for (int j = 0; j < 2; ++j) {
        const float bv = bias[(nt0 + j) * 16 + ln];
#pragma unroll
        for (int r = 0; r < 8; ++r) c[j][r] = bv;
    }

    const _Float16* w0  = Wp + (((size_t)(nt0    ) * KT) << 9) + ((size_t)lane << 4);
    const _Float16* w1  = Wp + (((size_t)(nt0 + 1) * KT) << 9) + ((size_t)lane << 4);
    const _Float16* sAl = &sA[lane * 16];

#pragma unroll 4
    for (int kt = 0; kt < KT; ++kt) {
        const v16h a  = *(const v16h*)(sAl + ((size_t)kt << 9));  // LDS, 32B
        const v16h b0 = *(const v16h*)(w0  + ((size_t)kt << 9));  // global, 32B
        const v16h b1 = *(const v16h*)(w1  + ((size_t)kt << 9));
        c[0] = __builtin_amdgcn_wmma_f32_16x16x32_f16(
                   false, a, false, b0, (short)0, c[0], false, false);
        c[1] = __builtin_amdgcn_wmma_f32_16x16x32_f16(
                   false, a, false, b1, (short)0, c[1], false, false);
    }

    // ---- LIF update. C/D layout: VGPR r -> row m = r + 8*half, col n = lane&15.
    float spkCnt = 0.0f;
#pragma unroll
    for (int j = 0; j < 2; ++j) {
        const int nG = (nt0 + j) * 16 + ln;
#pragma unroll
        for (int r = 0; r < 8; ++r) {
            const int row = mt * 16 + r + 8 * half;
            const size_t idx = (size_t)row * H + nG;
            float mv    = mem[idx];
            float reset = (mv > 1.0f) ? 1.0f : 0.0f;   // spike(mem_prev - thr)
            mv = beta * mv + c[j][r] - reset;          // thr = 1
            float s = (mv > 1.0f) ? 1.0f : 0.0f;       // spike(mem - thr)
            mem[idx] = mv;
            if constexpr (HAS_SPK) spkOut[idx] = (_Float16)s;
            if constexpr (HAS_SUM) memSum[idx] += mv;
            spkCnt += s;
        }
    }
    atomicAdd(spikeAcc, spkCnt);
}

// out[b,o] = (m3sum[b,:]/T) . Wr[o,:] + br[o]   (256x2 tiny readout, f32)
__global__ void snn_readout(const float* __restrict__ m3sum,
                            const float* __restrict__ Wr,
                            const float* __restrict__ br,
                            float* __restrict__ out)
{
    int tid = blockIdx.x * blockDim.x + threadIdx.x;
    if (tid >= BATCH * 2) return;
    int b = tid >> 1, o = tid & 1;
    const float* ms = m3sum + (size_t)b * H3;
    const float* w  = Wr + (size_t)o * H3;
    float acc = 0.0f;
    for (int h = 0; h < H3; ++h) acc += ms[h] * w[h];
    out[tid] = acc * (1.0f / (float)T_STEPS) + br[o];
}

__global__ void snn_stats(const float* __restrict__ spikeAcc,
                          float* __restrict__ out)
{
    int i = threadIdx.x;
    if (i >= 3) return;
    float denom = (i == 2) ? (float)((long long)T_STEPS * BATCH * H3)
                           : (float)((long long)T_STEPS * BATCH * H1);
    out[BATCH * 2 + i] = spikeAcc[i] / denom;
}

// ---------------------------------------------------------------------------
extern "C" void kernel_launch(void* const* d_in, const int* in_sizes, int n_in,
                              void* d_out, int out_size, void* d_ws, size_t ws_size,
                              hipStream_t stream) {
    (void)in_sizes; (void)n_in; (void)out_size; (void)ws_size;

    const float* spikes = (const float*)d_in[0];  // [200,256,512]
    const float* W1 = (const float*)d_in[1];      // [1024,512]
    const float* b1 = (const float*)d_in[2];
    const float* W2 = (const float*)d_in[3];      // [1024,1024]
    const float* b2 = (const float*)d_in[4];
    const float* W3 = (const float*)d_in[5];      // [512,1024]
    const float* b3 = (const float*)d_in[6];
    const float* Wr = (const float*)d_in[7];      // [2,512]
    const float* br = (const float*)d_in[8];
    float* out = (float*)d_out;                   // 515 f32

    // ---- workspace layout (all offsets 256B aligned) ----
    char* ws = (char*)d_ws;
    size_t off = 0;
    auto carve = [&](size_t bytes) -> char* {
        char* p = ws + off;
        off += (bytes + 255) & ~(size_t)255;
        return p;
    };
    // zero-initialized state region first (one memset covers it all)
    float*    m1    = (float*)carve((size_t)BATCH * H1 * 4);
    float*    m2    = (float*)carve((size_t)BATCH * H2 * 4);
    float*    m3    = (float*)carve((size_t)BATCH * H3 * 4);
    float*    m3sum = (float*)carve((size_t)BATCH * H3 * 4);
    float*    sAcc  = (float*)carve(3 * 4);
    size_t zeroBytes = off;
    // non-zeroed region
    _Float16* W1p  = (_Float16*)carve((size_t)H1 * IN_F * 2);
    _Float16* W2p  = (_Float16*)carve((size_t)H2 * H1 * 2);
    _Float16* W3p  = (_Float16*)carve((size_t)H3 * H2 * 2);
    _Float16* spk1 = (_Float16*)carve((size_t)BATCH * H1 * 2);
    _Float16* spk2 = (_Float16*)carve((size_t)BATCH * H2 * 2);

    hipMemsetAsync(ws, 0, zeroBytes, stream);

    // ---- pack weights into WMMA B-fragment layout (f16) ----
    snn_pack_w<<<(H1 * IN_F + 255) / 256, 256, 0, stream>>>(W1, W1p, H1, IN_F);
    snn_pack_w<<<(H2 * H1   + 255) / 256, 256, 0, stream>>>(W2, W2p, H2, H1);
    snn_pack_w<<<(H3 * H2   + 255) / 256, 256, 0, stream>>>(W3, W3p, H3, H2);

    // ---- 200-step scan: 3 fused GEMM+LIF layers per step ----
    // blocks = (BATCH/16) * (H/32) / 8
    const int blk1 = (BATCH / 16) * (H1 / 32) / 8;  // 64
    const int blk2 = (BATCH / 16) * (H2 / 32) / 8;  // 64
    const int blk3 = (BATCH / 16) * (H3 / 32) / 8;  // 32
    for (int t = 0; t < T_STEPS; ++t) {
        const float* xt = spikes + (size_t)t * BATCH * IN_F;
        snn_lif<true,  IN_F, H1, true,  false><<<blk1, 256, 0, stream>>>(
            xt,   W1p, b1, m1, spk1,    nullptr, sAcc + 0, BETA1);
        snn_lif<false, H1,   H2, true,  false><<<blk2, 256, 0, stream>>>(
            spk1, W2p, b2, m2, spk2,    nullptr, sAcc + 1, BETA2);
        snn_lif<false, H2,   H3, false, true ><<<blk3, 256, 0, stream>>>(
            spk2, W3p, b3, m3, nullptr, m3sum,   sAcc + 2, BETA3);
    }

    // ---- readout + stats ----
    snn_readout<<<2, 256, 0, stream>>>(m3sum, Wr, br, out);
    snn_stats<<<1, 4, 0, stream>>>(sAcc, out);
}
